// DeformConv_75273596829839
// MI455X (gfx1250) — matrix operand order repaired and checked by reading, not compiled, not run on hardware
//
#include <hip/hip_runtime.h>
#include <hip/hip_bf16.h>
#include <stdint.h>

typedef __attribute__((ext_vector_type(16))) _Float16 v16h;
typedef __attribute__((ext_vector_type(8)))  _Float16 v8h;
typedef __attribute__((ext_vector_type(8)))  float    v8f;
typedef int v4i_ __attribute__((vector_size(16)));   // b128 payload type for async builtins

#define BATCH 4
#define CIN   256
#define HH    56
#define WW    56
#define COUT  256
#define KK    9
#define HW    (HH*WW)            // 3136
#define NTOT  (BATCH*HW)         // 12544
#define KTOT  (CIN*KK)           // 2304
#define OFFC  (2*KK)             // 18
#define NKT   (KTOT/32)          // 72 K-steps

// ---------------------------------------------------------------------------
// CDNA5 async global->LDS copy (16B per lane), ASYNCcnt-tracked.
// ---------------------------------------------------------------------------
__device__ __forceinline__ void async_cp16(const _Float16* gsrc, _Float16* ldst) {
#if __has_builtin(__builtin_amdgcn_global_load_async_to_lds_b128)
    __builtin_amdgcn_global_load_async_to_lds_b128(
        (__attribute__((address_space(1))) v4i_*)(void*)gsrc,
        (__attribute__((address_space(3))) v4i_*)(void*)ldst, 0, 0);
#else
    uint32_t lofs = (uint32_t)(uintptr_t)(__attribute__((address_space(3))) void*)ldst;
    asm volatile("global_load_async_to_lds_b128 %0, %1, off"
                 :: "v"(lofs), "v"(gsrc) : "memory");
#endif
}

#if __has_builtin(__builtin_amdgcn_s_wait_asynccnt)
#define WAIT_ASYNC(n) __builtin_amdgcn_s_wait_asynccnt(n)
#else
#define WAIT_ASYNC(n) asm volatile("s_wait_asynccnt %0" :: "i"(n) : "memory")
#endif

// ---------------------------------------------------------------------------
// Kernel 1: offset-predicting 3x3 conv (f32, direct)
// off[b, c, h, w], c = kk*2 + d  (d=0 -> dy, d=1 -> dx)
// ---------------------------------------------------------------------------
__global__ __launch_bounds__(256) void dc_offset_conv(
    const float* __restrict__ x, const float* __restrict__ off_w,
    const float* __restrict__ off_b, float* __restrict__ off)
{
    int idx = blockIdx.x * 256 + threadIdx.x;
    if (idx >= BATCH * OFFC * HW) return;
    int hw = idx % HW;
    int oc = (idx / HW) % OFFC;
    int b  = idx / (OFFC * HW);
    int ho = hw / WW, wo = hw % WW;

    float acc = off_b[oc];
    const float* xb = x + (size_t)b * CIN * HW;
    const float* wb = off_w + (size_t)oc * CIN * 9;
    for (int ci = 0; ci < CIN; ++ci) {
        const float* xp = xb + ci * HW;
        const float* wp = wb + ci * 9;
        #pragma unroll
        for (int ky = 0; ky < 3; ++ky) {
            int iy = ho + ky - 1;
            if (iy < 0 || iy >= HH) continue;
            #pragma unroll
            for (int kx = 0; kx < 3; ++kx) {
                int ix = wo + kx - 1;
                if (ix < 0 || ix >= WW) continue;
                acc = fmaf(xp[iy * WW + ix], wp[ky * 3 + kx], acc);
            }
        }
    }
    off[idx] = acc;
}

// ---------------------------------------------------------------------------
// Kernel 2: repack weight [COUT][CIN][3][3] f32 -> Wh[o][k] f16, k = kk*256+cin
// ---------------------------------------------------------------------------
__global__ __launch_bounds__(256) void dc_weight_repack(
    const float* __restrict__ w, _Float16* __restrict__ wh)
{
    int idx = blockIdx.x * 256 + threadIdx.x;
    if (idx >= COUT * KTOT) return;
    int k = idx % KTOT;
    int o = idx / KTOT;
    int kk = k / CIN, ci = k % CIN;
    wh[idx] = (_Float16)w[(size_t)o * KTOT + ci * KK + kk];
}

// ---------------------------------------------------------------------------
// Kernel 3: bilinear sampling -> S[n][k] f16, n = b*HW + h*W + w, k = kk*256+cin
// grid (NTOT, KK), block 256 (one thread per cin)
// ---------------------------------------------------------------------------
__global__ __launch_bounds__(256) void dc_sample(
    const float* __restrict__ x, const float* __restrict__ off,
    _Float16* __restrict__ S)
{
    int n  = blockIdx.x;
    int kk = blockIdx.y;
    int ci = threadIdx.x;

    int hw = n % HW;
    int b  = n / HW;
    int ho = hw / WW, wo = hw % WW;

    const float* ob = off + ((size_t)b * OFFC + kk * 2) * HW + hw;
    float dy = ob[0];
    float dx = ob[HW];

    float py = (float)(ho - 1 + kk / 3) + dy;
    float px = (float)(wo - 1 + kk % 3) + dx;

    float fy = floorf(py), fx = floorf(px);
    int y0 = (int)fy, x0 = (int)fx;
    float ty = py - fy, tx = px - fx;
    int y1 = y0 + 1, x1 = x0 + 1;

    bool vy0 = (y0 >= 0) & (y0 < HH);
    bool vy1 = (y1 >= 0) & (y1 < HH);
    bool vx0 = (x0 >= 0) & (x0 < WW);
    bool vx1 = (x1 >= 0) & (x1 < WW);

    float w00 = (vy0 & vx0) ? (1.f - ty) * (1.f - tx) : 0.f;
    float w01 = (vy0 & vx1) ? (1.f - ty) * tx         : 0.f;
    float w10 = (vy1 & vx0) ? ty * (1.f - tx)         : 0.f;
    float w11 = (vy1 & vx1) ? ty * tx                 : 0.f;

    const float* xp = x + ((size_t)b * CIN + ci) * HW;
    int cy0 = min(max(y0, 0), HH - 1), cy1 = min(max(y1, 0), HH - 1);
    int cx0 = min(max(x0, 0), WW - 1), cx1 = min(max(x1, 0), WW - 1);

    float v = 0.f;
    v = fmaf(xp[cy0 * WW + cx0], w00, v);
    v = fmaf(xp[cy0 * WW + cx1], w01, v);
    v = fmaf(xp[cy1 * WW + cx0], w10, v);
    v = fmaf(xp[cy1 * WW + cx1], w11, v);

    S[(size_t)n * KTOT + kk * CIN + ci] = (_Float16)v;
}

// ---------------------------------------------------------------------------
// Kernel 4: WMMA GEMM  out[m,n] = sum_k Wh[m][k] * S[n][k]
// block tile 128(M) x 128(N), 8 waves: wave (wm in 0..3, wn in 0..1) does 32x64
// K-step 32, double-buffered LDS filled by async global->LDS DMA.
// All dims divide exactly -> no predication, EXEC stays all-ones for WMMA.
// ---------------------------------------------------------------------------
__global__ __launch_bounds__(256) void dc_gemm_wmma(
    const _Float16* __restrict__ Wh, const _Float16* __restrict__ S,
    float* __restrict__ out)
{
    // [buf][A/B][128 rows * 32 halfs]  -> 2 * 16KB = 32KB
    __shared__ __align__(16) _Float16 lds[2][2][128 * 32];

    const int tn0  = blockIdx.x * 128;
    const int tm0  = blockIdx.y * 128;
    const int tid  = threadIdx.x;
    const int lane = tid & 31;
    const int wave = tid >> 5;
    const int wm   = wave & 3;   // M group of 32
    const int wn   = wave >> 2;  // N group of 64
    const int lr   = lane & 15;
    const bool hi  = lane >= 16;

    // this thread's two 16B staging chunks per tile: c = t*256 + tid
    const int c0 = tid,        r0 = c0 >> 2, cc0 = c0 & 3;
    const int c1 = tid + 256,  r1 = c1 >> 2, cc1 = c1 & 3;

    const char* Ab = (const char*)Wh + (size_t)tm0 * (KTOT * 2);
    const char* Bb = (const char*)S  + (size_t)tn0 * (KTOT * 2);

    v8f acc[2][4];
    #pragma unroll
    for (int i = 0; i < 2; ++i)
        #pragma unroll
        for (int j = 0; j < 4; ++j) {
            v8f z = {};
            acc[i][j] = z;
        }

    // issue stage 0 into buf 0 (4 async ops per wave)
    {
        const int kb = 0;
        async_cp16((const _Float16*)(Ab + (size_t)r0 * (KTOT*2) + kb + cc0*16), &lds[0][0][c0*8]);
        async_cp16((const _Float16*)(Ab + (size_t)r1 * (KTOT*2) + kb + cc1*16), &lds[0][0][c1*8]);
        async_cp16((const _Float16*)(Bb + (size_t)r0 * (KTOT*2) + kb + cc0*16), &lds[0][1][c0*8]);
        async_cp16((const _Float16*)(Bb + (size_t)r1 * (KTOT*2) + kb + cc1*16), &lds[0][1][c1*8]);
    }

    for (int kt = 0; kt < NKT; ++kt) {
        const int cur = kt & 1;
        const int nxt = cur ^ 1;

        if (kt + 1 < NKT) {
            // prefetch next stage into the other buffer (consumed at kt-1, safe
            // because iteration kt-1 ended with a barrier)
            const int kb = (kt + 1) * 64;   // byte offset along K
            async_cp16((const _Float16*)(Ab + (size_t)r0 * (KTOT*2) + kb + cc0*16), &lds[nxt][0][c0*8]);
            async_cp16((const _Float16*)(Ab + (size_t)r1 * (KTOT*2) + kb + cc1*16), &lds[nxt][0][c1*8]);
            async_cp16((const _Float16*)(Bb + (size_t)r0 * (KTOT*2) + kb + cc0*16), &lds[nxt][1][c0*8]);
            async_cp16((const _Float16*)(Bb + (size_t)r1 * (KTOT*2) + kb + cc1*16), &lds[nxt][1][c1*8]);
            WAIT_ASYNC(4);   // 4 outstanding = next stage; current stage resident
        } else {
            WAIT_ASYNC(0);
        }
        __syncthreads();     // all waves' current-stage DMA visible

        const _Float16* Al = lds[cur][0];
        const _Float16* Bl = lds[cur][1];

        // A fragments: row M = lane%16; lane<16 holds K{0..7,16..23}, lane>=16 K{8..15,24..31}
        v16h a[2];
        #pragma unroll
        for (int i = 0; i < 2; ++i) {
            int m = wm * 32 + i * 16 + lr;
            const _Float16* base = &Al[m * 32 + (hi ? 8 : 0)];
            v8h lo = *(const v8h*)(base);
            v8h hh = *(const v8h*)(base + 16);
            a[i] = __builtin_shufflevector(lo, hh, 0,1,2,3,4,5,6,7,8,9,10,11,12,13,14,15);
        }
        // B fragments: col N = lane%16; lane<16 holds K{0..7,16..23}, lane>=16 K{8..15,24..31}
        v16h b[4];
        #pragma unroll
        for (int j = 0; j < 4; ++j) {
            int n = wn * 64 + j * 16 + lr;
            const _Float16* base = &Bl[n * 32 + (hi ? 8 : 0)];
            v8h lo = *(const v8h*)(base);
            v8h hh = *(const v8h*)(base + 16);
            b[j] = __builtin_shufflevector(lo, hh, 0,1,2,3,4,5,6,7,8,9,10,11,12,13,14,15);
        }

        #pragma unroll
        for (int i = 0; i < 2; ++i)
            #pragma unroll
            for (int j = 0; j < 4; ++j)
                acc[i][j] = __builtin_amdgcn_wmma_f32_16x16x32_f16(
                    false, a[i], false, b[j], (short)0, acc[i][j], false, false);

        __syncthreads();     // done reading buf[cur]; safe to overwrite next iter
    }

    // store: C fragment VGPR r, lanes 0-15 -> M=r, lanes 16-31 -> M=r+8; N = lane%16
    #pragma unroll
    for (int j = 0; j < 4; ++j) {
        int n  = tn0 + wn * 64 + j * 16 + lr;
        int b_ = n / HW, hw = n % HW;
        float* op = out + (size_t)b_ * COUT * HW + hw;
        #pragma unroll
        for (int i = 0; i < 2; ++i) {
            int mbase = tm0 + wm * 32 + i * 16 + (hi ? 8 : 0);
            #pragma unroll
            for (int r = 0; r < 8; ++r)
                op[(size_t)(mbase + r) * HW] = acc[i][j][r];
        }
    }
}

// ---------------------------------------------------------------------------
// launcher
// ---------------------------------------------------------------------------
extern "C" void kernel_launch(void* const* d_in, const int* in_sizes, int n_in,
                              void* d_out, int out_size, void* d_ws, size_t ws_size,
                              hipStream_t stream) {
    const float* x      = (const float*)d_in[0];
    const float* weight = (const float*)d_in[1];
    const float* off_w  = (const float*)d_in[2];
    const float* off_b  = (const float*)d_in[3];
    float* out = (float*)d_out;

    // workspace layout (256B aligned):
    //   off : BATCH*18*HW f32        =   903,168 B at 0
    //   Wh  : COUT*KTOT  f16         = 1,179,648 B at 903,168
    //   S   : NTOT*KTOT  f16         = 57,802,752 B at 2,082,816
    char* ws = (char*)d_ws;
    float*    off = (float*)(ws);
    _Float16* Wh  = (_Float16*)(ws + 903168);
    _Float16* S   = (_Float16*)(ws + 2082816);

    {   // offset conv
        int total = BATCH * OFFC * HW;             // 225792
        dc_offset_conv<<<(total + 255) / 256, 256, 0, stream>>>(x, off_w, off_b, off);
    }
    {   // weight repack (independent)
        int total = COUT * KTOT;                   // 589824
        dc_weight_repack<<<(total + 255) / 256, 256, 0, stream>>>(weight, Wh);
    }
    {   // bilinear sampling
        dim3 grid(NTOT, KK);
        dc_sample<<<grid, 256, 0, stream>>>(x, off, S);
    }
    {   // WMMA implicit GEMM: N tiles = 12544/128 = 98, M tiles = 256/128 = 2
        dim3 grid(NTOT / 128, COUT / 128);
        dc_gemm_wmma<<<grid, 256, 0, stream>>>(Wh, S, out);
    }
}
